// AlexNetOWT_BN_brevitas_71648644431999
// MI455X (gfx1250) — matrix operand order repaired and checked by reading, not compiled, not run on hardware
//
#include <hip/hip_runtime.h>
#include <hip/hip_bf16.h>
#include <cstdint>

typedef __attribute__((ext_vector_type(16))) _Float16 v16h;
typedef __attribute__((ext_vector_type(8)))  _Float16 v8h;
typedef __attribute__((ext_vector_type(8)))  float    v8f;

#define WSCALE 0.1f
#define BN_EPS 1e-5f

// ---------------------------------------------------------------------------
// bn0: per-channel affine on f32 input, cast to f16 (strided row output)
// ---------------------------------------------------------------------------
__global__ __launch_bounds__(256) void bn_affine_f32_to_f16(
    const float* __restrict__ X, _Float16* __restrict__ Y,
    const float* __restrict__ g, const float* __restrict__ bp,
    const float* __restrict__ m, const float* __restrict__ vv,
    int C, int L, int LS, size_t total)
{
    size_t i = (size_t)blockIdx.x * blockDim.x + threadIdx.x;
    if (i >= total) return;
    int l = (int)(i % (size_t)L);
    size_t t = i / (size_t)L;
    int c = (int)(t % (size_t)C);
    size_t b = t / (size_t)C;
    float inv = g[c] * rsqrtf(vv[c] + BN_EPS);
    float sh  = bp[c] - m[c] * inv;
    Y[(b * C + c) * (size_t)LS + l] = (_Float16)(X[i] * inv + sh);
}

// ---------------------------------------------------------------------------
// bn -> sign -> maxpool(p)  (sign∘bn commutes with max since bn scale > 0)
// ---------------------------------------------------------------------------
__global__ __launch_bounds__(256) void bn_sign_pool(
    const _Float16* __restrict__ X, _Float16* __restrict__ Y,
    const float* __restrict__ g, const float* __restrict__ bp,
    const float* __restrict__ m, const float* __restrict__ vv,
    int C, int LinS, int pool, int Lout, int LoutS, size_t total)
{
    size_t i = (size_t)blockIdx.x * blockDim.x + threadIdx.x;
    if (i >= total) return;
    int l = (int)(i % (size_t)Lout);
    size_t t = i / (size_t)Lout;
    int c = (int)(t % (size_t)C);
    size_t b = t / (size_t)C;
    const _Float16* xp = X + (b * C + c) * (size_t)LinS + (size_t)l * pool;
    float mx = -3.0e38f;
    for (int j = 0; j < pool; ++j) mx = fmaxf(mx, (float)xp[j]);
    float inv = g[c] * rsqrtf(vv[c] + BN_EPS);
    float y = mx * inv + (bp[c] - m[c] * inv);
    Y[(b * C + c) * (size_t)LoutS + l] = (_Float16)(y >= 0.f ? 1.f : -1.f);
}

// ---------------------------------------------------------------------------
// Weight prepack: binarize (±0.1) into exact WMMA A-fragment lane order.
// P[((ct*nChunk + ch)*32 + lane)*16 + j],  K ordering kk = t*Ci + ci.
// ---------------------------------------------------------------------------
__global__ __launch_bounds__(256) void pack_w(
    const float* __restrict__ W, _Float16* __restrict__ P,
    int Ci, int Co, int ksz, int nChunk, size_t total)
{
    size_t idx = (size_t)blockIdx.x * blockDim.x + threadIdx.x;
    if (idx >= total) return;
    int j    = (int)(idx & 15);
    int lane = (int)((idx >> 4) & 31);
    size_t tile = idx >> 9;
    int ch = (int)(tile % (size_t)nChunk);
    int ct = (int)(tile / (size_t)nChunk);
    int vi = j >> 1, lo = j & 1;
    int kk = ch * 32 + (lane >> 4) * 16 + (vi << 1) + lo;
    int co = ct * 16 + (lane & 15);
    int K = Ci * ksz;
    _Float16 val = (_Float16)0.f;
    if (kk < K && co < Co) {
        int t = kk / Ci, ci = kk - t * Ci;
        float w = W[((size_t)co * Ci + ci) * ksz + t];
        val = (_Float16)(w >= 0.f ? WSCALE : -WSCALE);
    }
    P[idx] = val;
}

// ---------------------------------------------------------------------------
// Implicit-GEMM dilated conv, binary weights, f16 WMMA, f32 accum.
// Stage 1: async global->LDS byte copy of [ci][spanR] tile (16B units).
// Stage 2: LDS transpose into [pos][ci] so a B fragment is one contiguous
//          32B-aligned v16h load (CI % 32 == 0 for all layers but conv0).
// Block 128 thr (4 waves), 64 output positions, ct = blockIdx.z*4 + wave.
// Row strides LinS/LoutS are multiples of 8 halves (16B) for async alignment.
// ---------------------------------------------------------------------------
template <int CI, int KSZ, int DIL>
__global__ __launch_bounds__(128) void conv_wmma(
    const _Float16* __restrict__ X, const _Float16* __restrict__ Apack,
    const float* __restrict__ bias, _Float16* __restrict__ Y,
    int Co, int Lout, int LinS, int LoutS)
{
    constexpr int K     = CI * KSZ;
    constexpr int NCH   = (K + 31) / 32;
    constexpr int SPAN  = 64 + (KSZ - 1) * DIL;
    constexpr int UNITS = (SPAN * 2 + 15) / 16;      // 16B copy units per row
    constexpr int SPANR = UNITS * 8;                 // raw row stride (halves)
    constexpr int CIP   = (CI % 32 == 0) ? (CI + 16) : CI;
    __shared__ __align__(16) _Float16 raw[CI * SPANR];
    __shared__ __align__(16) _Float16 xs[SPAN * CIP];

    const int b   = blockIdx.y;
    const int l0  = blockIdx.x * 64;
    const int tid = threadIdx.x;

    // ---- stage 1: async contiguous copies, one 16B unit per lane ----
    {
        const char* base = (const char*)(X + (size_t)b * CI * LinS + l0);
        for (int u = tid; u < CI * UNITS; u += 128) {
            int ci = u / UNITS, un = u - ci * UNITS;
            uint64_t g = (uint64_t)(uintptr_t)(base +
                         ((size_t)ci * LinS * 2) + (size_t)un * 16);
            uint32_t d = (uint32_t)(uintptr_t)(&raw[ci * SPANR + un * 8]);
            asm volatile("global_load_async_to_lds_b128 %0, %1, off"
                         :: "v"(d), "v"(g) : "memory");
        }
        asm volatile("s_wait_asynccnt 0" ::: "memory");
    }
    __syncthreads();

    // ---- stage 2: LDS transpose raw[ci][p] -> xs[p][ci] ----
    if constexpr (CI % 8 == 0) {
        constexpr int NC8 = CI / 8;
        for (int i = tid; i < SPAN * NC8; i += 128) {
            int p = i / NC8, c8 = i - p * NC8;
            v8h t;
            #pragma unroll
            for (int j = 0; j < 8; ++j) t[j] = raw[(c8 * 8 + j) * SPANR + p];
            *(v8h*)&xs[p * CIP + c8 * 8] = t;
        }
    } else {
        for (int i = tid; i < SPAN * CI; i += 128) {
            int ci = i % CI, p = i / CI;
            xs[p * CIP + ci] = raw[ci * SPANR + p];
        }
    }
    __syncthreads();

    const int wave = tid >> 5;
    const int lane = tid & 31;
    const int nlo  = lane & 15;
    const int hi   = lane >> 4;
    const int ct   = blockIdx.z * 4 + wave;

    v8f acc[4];
    #pragma unroll
    for (int s = 0; s < 4; ++s) acc[s] = (v8f){0,0,0,0,0,0,0,0};

    const v16h* Ap = (const v16h*)Apack;
    size_t abase = (size_t)ct * NCH * 32 + lane;

    for (int ch = 0; ch < NCH; ++ch) {
        __builtin_prefetch((const void*)(Ap + abase + (size_t)(ch + 8) * 32), 0, 1);
        v16h a = Ap[abase + (size_t)ch * 32];
        if constexpr (CI % 32 == 0) {
            const int t  = (ch * 32) / CI;           // compile-time shifts
            const int c0 = (ch * 32) % CI;
            #pragma unroll
            for (int s = 0; s < 4; ++s) {
                const v16h bf = *(const v16h*)
                    &xs[(s * 16 + nlo + t * DIL) * CIP + c0 + hi * 16];
                acc[s] = __builtin_amdgcn_wmma_f32_16x16x32_f16(
                    false, a, false, bf, (short)0, acc[s], false, false);
            }
        } else {                                     // conv0 fallback (CI==3)
            const int kk0 = ch * 32;
            #pragma unroll
            for (int s = 0; s < 4; ++s) {
                v16h bf;
                #pragma unroll
                for (int vi = 0; vi < 8; ++vi) {
                    int kk = kk0 + hi * 16 + 2 * vi;
                    _Float16 h0 = (_Float16)0.f, h1 = (_Float16)0.f;
                    if (kk < K) {
                        int t = kk / CI, ci = kk - t * CI;
                        h0 = xs[(s * 16 + nlo + t * DIL) * CIP + ci];
                    }
                    if (kk + 1 < K) {
                        int k1 = kk + 1, t = k1 / CI, ci = k1 - t * CI;
                        h1 = xs[(s * 16 + nlo + t * DIL) * CIP + ci];
                    }
                    bf[2 * vi] = h0; bf[2 * vi + 1] = h1;
                }
                acc[s] = __builtin_amdgcn_wmma_f32_16x16x32_f16(
                    false, a, false, bf, (short)0, acc[s], false, false);
            }
        }
    }

    // epilogue: bias + f16 store (C/D: M = r + 8*hi, N = lane&15)
    #pragma unroll
    for (int s = 0; s < 4; ++s) {
        int pos = l0 + s * 16 + nlo;
        if (pos < Lout) {
            #pragma unroll
            for (int r = 0; r < 8; ++r) {
                int co = ct * 16 + r + 8 * hi;
                float val = acc[s][r] + bias[co];
                Y[((size_t)b * Co + co) * (size_t)LoutS + pos] = (_Float16)val;
            }
        }
    }
}

// ---------------------------------------------------------------------------
// FC GEMM: out[b,o] = sum_k A[b,k] * sign(W[o,k])*0.1,  M = 16 (batch).
// 8 waves split K per 16-output tile, LDS reduce, fused bn epilogue.
// mode 0: bn+sign -> f16 [16,N];  mode 1: bn -> f32 [16,N].
// ---------------------------------------------------------------------------
__global__ __launch_bounds__(256) void fc_wmma(
    const _Float16* __restrict__ A, const float* __restrict__ W,
    const float* __restrict__ g, const float* __restrict__ bp,
    const float* __restrict__ m, const float* __restrict__ vv,
    void* __restrict__ outp, int K, int N, int mode)
{
    __shared__ float red[8 * 32 * 8];
    const int o0   = blockIdx.x * 16;
    const int tid  = threadIdx.x;
    const int wave = tid >> 5;
    const int lane = tid & 31;
    const int nlo  = lane & 15;
    const int hi   = lane >> 4;

    v8f acc = (v8f){0,0,0,0,0,0,0,0};
    const int ob = o0 + nlo;
    const bool ov = ob < N;
    const _Float16* Arow = A + (size_t)nlo * K;
    const float* Wrow = ov ? (W + (size_t)ob * K) : W;

    for (int kk0 = wave * 32; kk0 < K; kk0 += 8 * 32) {
        // A fragment = two contiguous 16B runs: kk0+hi*8+{0..7}, +16+{0..7}
        v8h a_lo = *(const v8h*)(Arow + kk0 + hi * 8);
        v8h a_hi = *(const v8h*)(Arow + kk0 + hi * 8 + 16);
        v16h a = __builtin_shufflevector(a_lo, a_hi,
                 0,1,2,3,4,5,6,7,8,9,10,11,12,13,14,15);
        const float* Wr = Wrow + kk0 + hi * 8;
        __builtin_prefetch((const void*)(Wr + 8 * 32), 0, 1);
        v16h bf;
        if (ov) {
            #pragma unroll
            for (int q = 0; q < 16; ++q)
                bf[q] = (_Float16)(Wr[q + (q >= 8 ? 8 : 0)] >= 0.f ? WSCALE : -WSCALE);
        } else {
            #pragma unroll
            for (int q = 0; q < 16; ++q) bf[q] = (_Float16)0.f;
        }
        acc = __builtin_amdgcn_wmma_f32_16x16x32_f16(
            false, a, false, bf, (short)0, acc, false, false);
    }

    #pragma unroll
    for (int r = 0; r < 8; ++r) red[(wave * 32 + lane) * 8 + r] = acc[r];
    __syncthreads();

    if (wave == 0) {
        float s[8];
        #pragma unroll
        for (int r = 0; r < 8; ++r) s[r] = 0.f;
        for (int w = 0; w < 8; ++w)
            #pragma unroll
            for (int r = 0; r < 8; ++r) s[r] += red[(w * 32 + lane) * 8 + r];

        float inv = 0.f, sh = 0.f;
        if (ov) { inv = g[ob] * rsqrtf(vv[ob] + BN_EPS); sh = bp[ob] - m[ob] * inv; }
        #pragma unroll
        for (int r = 0; r < 8; ++r) {
            int bb = r + 8 * hi;
            float y = s[r] * inv + sh;
            if (ov) {
                if (mode == 0)
                    ((_Float16*)outp)[(size_t)bb * N + ob] =
                        (_Float16)(y >= 0.f ? 1.f : -1.f);
                else
                    ((float*)outp)[(size_t)bb * N + ob] = y;
            }
        }
    }
}

// ---------------------------------------------------------------------------
extern "C" void kernel_launch(void* const* d_in, const int* in_sizes, int n_in,
                              void* d_out, int out_size, void* d_ws, size_t ws_size,
                              hipStream_t stream) {
    (void)in_sizes; (void)n_in; (void)out_size; (void)ws_size;
    const float* x = (const float*)d_in[0];
    const float* convW[12]; const float* convB[12];
    for (int i = 0; i < 12; ++i) {
        convW[i] = (const float*)d_in[1 + 2 * i];
        convB[i] = (const float*)d_in[2 + 2 * i];
    }
    const float *bnG[6], *bnB[6], *bnM[6], *bnV[6];
    for (int i = 0; i < 6; ++i) {
        bnG[i] = (const float*)d_in[25 + 4 * i + 0];
        bnB[i] = (const float*)d_in[25 + 4 * i + 1];
        bnM[i] = (const float*)d_in[25 + 4 * i + 2];
        bnV[i] = (const float*)d_in[25 + 4 * i + 3];
    }
    const float* fc1w = (const float*)d_in[49];
    const float* fc2w = (const float*)d_in[50];
    const float *f1G = (const float*)d_in[51], *f1B = (const float*)d_in[52],
                *f1M = (const float*)d_in[53], *f1V = (const float*)d_in[54];
    const float *f2G = (const float*)d_in[55], *f2B = (const float*)d_in[56],
                *f2M = (const float*)d_in[57], *f2V = (const float*)d_in[58];

    // workspace: two f16 ping-pong activation buffers + packed weights
    _Float16* buf0 = (_Float16*)d_ws;                        // <= 32,358,400 B
    _Float16* buf1 = (_Float16*)((char*)d_ws + 32358400);    // <= 32,620,544 B

    static const int ciA[12] = {3,64,64,64,64,256,256,256,256,256,128,128};
    static const int coA[12] = {64,64,64,64,256,256,256,256,256,128,128,128};
    static const int kA[12]  = {67,64,4,3,4,3,3,3,3,3,3,3};
    _Float16* pk[12];
    size_t off = 64978944;                                   // after buf0+buf1
    for (int i = 0; i < 12; ++i) {
        int K = ciA[i] * kA[i];
        int nch = (K + 31) / 32;
        pk[i] = (_Float16*)((char*)d_ws + off);
        size_t elems = (size_t)(coA[i] / 16) * nch * 512;
        off += elems * sizeof(_Float16);
        pack_w<<<(unsigned)((elems + 255) / 256), 256, 0, stream>>>(
            convW[i], pk[i], ciA[i], coA[i], kA[i], nch, elems);
    }

    // bn0 (f32 -> f16), stride 15992 (already 16B aligned)
    {
        size_t tot = 16ull * 3 * 15992;
        bn_affine_f32_to_f16<<<(unsigned)((tot + 255) / 256), 256, 0, stream>>>(
            x, buf0, bnG[0], bnB[0], bnM[0], bnV[0], 3, 15992, 15992, tot);
    }

    auto pool = [&](const _Float16* in, _Float16* out, int bi, int C,
                    int LinS, int p, int Lout, int LoutS) {
        size_t tot = 16ull * C * Lout;
        bn_sign_pool<<<(unsigned)((tot + 255) / 256), 256, 0, stream>>>(
            in, out, bnG[bi], bnB[bi], bnM[bi], bnV[bi],
            C, LinS, p, Lout, LoutS, tot);
    };
    // row strides padded to multiples of 8 halves (16B) for async staging
    #define CONV(i, CIc, KSZc, DILc, in, out, CO, LOUT, LINS, LOUTS)             \
        conv_wmma<CIc, KSZc, DILc>                                               \
            <<<dim3(((LOUT) + 63) / 64, 16, (CO) / 64), 128, 0, stream>>>(       \
                in, pk[i], convB[i], out, CO, LOUT, LINS, LOUTS)

    CONV(0,   3, 67, 1, buf0, buf1,  64, 15926, 15992, 15928);
    CONV(1,  64, 64, 2, buf1, buf0,  64, 15800, 15928, 15800);
    pool(buf0, buf1, 1,  64, 15800, 4, 3950, 3952);
    CONV(2,  64,  4, 2, buf1, buf0,  64,  3944,  3952,  3944);
    CONV(3,  64,  3, 5, buf0, buf1,  64,  3934,  3944,  3936);
    pool(buf1, buf0, 2,  64,  3936, 2, 1967, 1968);
    CONV(4,  64,  4, 1, buf0, buf1, 256,  1964,  1968,  1968);
    CONV(5, 256,  3, 2, buf1, buf0, 256,  1960,  1968,  1960);
    pool(buf0, buf1, 3, 256,  1960, 2,  980,  984);
    CONV(6, 256,  3, 1, buf1, buf0, 256,   978,   984,   984);
    CONV(7, 256,  3, 2, buf0, buf1, 256,   974,   984,   976);
    CONV(8, 256,  3, 5, buf1, buf0, 256,   964,   976,   968);
    pool(buf0, buf1, 4, 256,   968, 2,  482,  488);
    CONV(9, 256,  3, 1, buf1, buf0, 128,   480,   488,   480);
    CONV(10,128,  3, 2, buf0, buf1, 128,   476,   480,   480);
    CONV(11,128,  3, 5, buf1, buf0, 128,   466,   480,   472);
    // bn5 + sign only; output CONTIGUOUS (stride 466) for the fc1 reshape
    pool(buf0, buf1, 5, 128,   472, 1,  466,  466);
    #undef CONV

    // fc1: [16,59648] x [4096,59648]^T (+bnf1+sign) -> f16 [16,4096]
    fc_wmma<<<4096 / 16, 256, 0, stream>>>(buf1, fc1w, f1G, f1B, f1M, f1V,
                                           (void*)buf0, 128 * 466, 4096, 0);
    // fc2: [16,4096] x [1000,4096]^T (+bnf2) -> f32 d_out [16,1000]
    fc_wmma<<<(1000 + 15) / 16, 256, 0, stream>>>(buf0, fc2w, f2G, f2B, f2M, f2V,
                                                  d_out, 4096, 1000, 1);
}
